// ExpertChoiceRouter_55508157334139
// MI455X (gfx1250) — matrix operand order repaired and checked by reading, not compiled, not run on hardware
//
#include <hip/hip_runtime.h>
#include <hip/hip_bf16.h>

// ---------------- problem constants ----------------
#define Bn 4
#define Tn 8192
#define Dn 512
#define En 8
#define Hn 2048
#define Cn 1024

typedef __attribute__((ext_vector_type(16))) __bf16 v16bf;
typedef __attribute__((ext_vector_type(8)))  float  v8f;
typedef int v4i_ __attribute__((vector_size(16)));   // matches builtin's int4 pointee

// ---- CDNA5 async global->LDS copy (ASYNCcnt path), with safe fallback ----
#if defined(__has_builtin)
#if __has_builtin(__builtin_amdgcn_global_load_async_to_lds_b128) && \
    __has_builtin(__builtin_amdgcn_s_wait_asynccnt)
#define HAVE_ASYNC_LDS 1
#endif
#endif
#ifndef HAVE_ASYNC_LDS
#define HAVE_ASYNC_LDS 0
#endif

__device__ __forceinline__ void async_copy16(const unsigned short* gsrc, unsigned short* ldst) {
#if HAVE_ASYNC_LDS
  __builtin_amdgcn_global_load_async_to_lds_b128(
      (__attribute__((address_space(1))) v4i_*)gsrc,
      (__attribute__((address_space(3))) v4i_*)ldst,
      0, 0);
#else
  *(uint4*)ldst = *(const uint4*)gsrc;
#endif
}
__device__ __forceinline__ void async_join() {
#if HAVE_ASYNC_LDS
  __builtin_amdgcn_s_wait_asynccnt(0);
#endif
}

// ---- fast tanh: prefer CDNA5 v_tanh_f32 transcendental ----
__device__ __forceinline__ float fast_tanh(float z) {
#if defined(__has_builtin)
#if __has_builtin(__builtin_amdgcn_tanhf)
  return __builtin_amdgcn_tanhf(z);
#elif __has_builtin(__builtin_amdgcn_tanh_f32)
  return __builtin_amdgcn_tanh_f32(z);
#else
  float e = __expf(2.0f * z);
  return 1.0f - 2.0f / (e + 1.0f);
#endif
#else
  float e = __expf(2.0f * z);
  return 1.0f - 2.0f / (e + 1.0f);
#endif
}
// tanh-form GELU (error ~1e-3, below bf16 quantization of h)
__device__ __forceinline__ float gelu_f(float v) {
  float inner = 0.7978845608f * (v + 0.044715f * v * v * v);
  return 0.5f * v * (1.0f + fast_tanh(inner));
}

__device__ __forceinline__ unsigned short f2bf(float f) {
  unsigned u = __float_as_uint(f);
  unsigned r = u + 0x7FFFu + ((u >> 16) & 1u);   // round-to-nearest-even
  return (unsigned short)(r >> 16);
}

// A-fragment (16x32 bf16, MxK): lane holds row m=lane&15; element j -> K = (lane>>4)*16 + j
__device__ __forceinline__ v16bf load_a_frag(const unsigned short* p /* row base + k0 + half*16 */) {
  union { v16bf v; uint4 q[2]; } u;
  u.q[0] = *(const uint4*)(p);
  u.q[1] = *(const uint4*)(p + 8);
  return u.v;
}
// B-fragment (32x16 bf16, KxN) from a K-contiguous (transposed) row:
// lane holds col n=lane&15; element j -> K = (lane>>4)*16 + j  (16 contiguous K values)
__device__ __forceinline__ v16bf load_b_frag(const unsigned short* p /* row base + k0 + half*16 */) {
  union { v16bf v; uint4 q[2]; } u;
  u.q[0] = *(const uint4*)(p);
  u.q[1] = *(const uint4*)(p + 8);
  return u.v;
}

// ---------------- kernel 1: zero output + norm ----------------
__global__ void k_init(float* __restrict__ out, float* __restrict__ norm) {
  size_t stride = (size_t)gridDim.x * blockDim.x;
  for (size_t i = (size_t)blockIdx.x * blockDim.x + threadIdx.x; i < (size_t)Bn * Tn * Dn; i += stride)
    out[i] = 0.0f;
  for (size_t i = (size_t)blockIdx.x * blockDim.x + threadIdx.x; i < (size_t)Bn * Tn; i += stride)
    norm[i] = 0.0f;
}

// ---------------- kernel 2: gating scores, transposed to (B,E,T) ----------------
__global__ __launch_bounds__(256) void k_gate(const float* __restrict__ x,
                                              const float* __restrict__ Wg,
                                              float* __restrict__ probs) {
  __shared__ float sWg[Dn * En];                     // 16 KB
  for (int i = threadIdx.x; i < Dn * En; i += 256) sWg[i] = Wg[i];
  __syncthreads();
  const int wave = threadIdx.x >> 5, lane = threadIdx.x & 31;
  const int tok = blockIdx.x * 8 + wave;             // [0, B*T)
  const int b = tok >> 13, t = tok & (Tn - 1);
  const float* xp = x + (size_t)tok * Dn;
  float acc[En];
#pragma unroll
  for (int e = 0; e < En; ++e) acc[e] = 0.0f;
  for (int d = lane; d < Dn; d += 32) {
    float xv = xp[d];
    const float* wg = &sWg[d * En];
#pragma unroll
    for (int e = 0; e < En; ++e) acc[e] += xv * wg[e];
  }
#pragma unroll
  for (int e = 0; e < En; ++e) {
    float v = acc[e];
    for (int off = 16; off >= 1; off >>= 1) v += __shfl_xor(v, off, 32);
    acc[e] = v;
  }
  if (lane == 0) {
#pragma unroll
    for (int e = 0; e < En; ++e)
      probs[((size_t)(b * En + e)) * Tn + t] = acc[e];
  }
}

// ---------------- kernel 3: softmax over T for each (b,e) row ----------------
__global__ __launch_bounds__(1024) void k_softmax(float* __restrict__ probs) {
  __shared__ float red[1024];
  float* row = probs + (size_t)blockIdx.x * Tn;
  const int tid = threadIdx.x;
  float mx = -1e30f;
  for (int i = tid; i < Tn; i += 1024) mx = fmaxf(mx, row[i]);
  red[tid] = mx; __syncthreads();
  for (int s = 512; s > 0; s >>= 1) { if (tid < s) red[tid] = fmaxf(red[tid], red[tid + s]); __syncthreads(); }
  mx = red[0]; __syncthreads();
  float sum = 0.0f;
  for (int i = tid; i < Tn; i += 1024) sum += __expf(row[i] - mx);
  red[tid] = sum; __syncthreads();
  for (int s = 512; s > 0; s >>= 1) { if (tid < s) red[tid] += red[tid + s]; __syncthreads(); }
  float inv = 1.0f / red[0]; __syncthreads();
  for (int i = tid; i < Tn; i += 1024) row[i] = __expf(row[i] - mx) * inv;
}

// ---------------- kernel 4: top-C radix select per (b,e) row ----------------
__global__ __launch_bounds__(256) void k_topc(const float* __restrict__ probs,
                                              int* __restrict__ topk_idx,
                                              float* __restrict__ topk_val,
                                              float* __restrict__ norm) {
  const int row = blockIdx.x;         // b*E + e
  const int b = row >> 3;
  const int tid = threadIdx.x;
  const float* p = probs + (size_t)row * Tn;
  __shared__ unsigned hist[256];
  __shared__ unsigned sSel, sAbove;
  __shared__ int posG, posE;

  unsigned prefix = 0;
  int remaining = Cn;
  for (int pass = 3; pass >= 0; --pass) {
    hist[tid] = 0;
    __syncthreads();
    unsigned mask = (pass == 3) ? 0u : (0xFFFFFFFFu << ((pass + 1) * 8));
    for (int i = tid; i < Tn; i += 256) {
      unsigned key = __float_as_uint(p[i]);     // softmax > 0 -> monotone bits
      if ((key & mask) == prefix) atomicAdd(&hist[(key >> (pass * 8)) & 255], 1u);
    }
    __syncthreads();
    if (tid == 0) {
      unsigned cum = 0, above = 0; int sel = 0;
      for (int bkt = 255; bkt >= 0; --bkt) {
        unsigned c = hist[bkt];
        if (cum + c >= (unsigned)remaining) { sel = bkt; above = cum; break; }
        cum += c;
      }
      sSel = (unsigned)sel; sAbove = above;
    }
    __syncthreads();
    prefix |= (sSel << (pass * 8));
    remaining -= (int)sAbove;
  }
  const unsigned threshold = prefix;            // key of the C-th largest
  if (tid == 0) { posG = 0; posE = Cn - remaining; }
  __syncthreads();
  for (int i = tid; i < Tn; i += 256) {
    float v = p[i];
    unsigned key = __float_as_uint(v);
    int slot = -1;
    if (key > threshold) slot = atomicAdd(&posG, 1);
    else if (key == threshold) { int s2 = atomicAdd(&posE, 1); if (s2 < Cn) slot = s2; }
    if (slot >= 0) {
      topk_idx[(size_t)row * Cn + slot] = i;
      topk_val[(size_t)row * Cn + slot] = v;
      unsafeAtomicAdd(&norm[(size_t)b * Tn + i], v);
    }
  }
}

// ---------------- kernel 5: convert + transpose weights to bf16 ----------------
// W1 [E][D][H] -> W1T [E][H][D] ; W2 [E][H][D] -> W2T [E][D][H]
__global__ void k_convW(const float* __restrict__ W1, const float* __restrict__ W2,
                        unsigned short* __restrict__ W1T, unsigned short* __restrict__ W2T) {
  const size_t total = (size_t)En * Dn * Hn;
  size_t stride = (size_t)gridDim.x * blockDim.x;
  for (size_t i = (size_t)blockIdx.x * blockDim.x + threadIdx.x; i < total; i += stride) {
    size_t e = i / ((size_t)Dn * Hn), r = i % ((size_t)Dn * Hn);
    size_t h = r / Dn, d = r % Dn;                       // W1T[e][h][d]
    W1T[i] = f2bf(W1[(e * Dn + d) * Hn + h]);
    size_t d2 = r / Hn, h2 = r % Hn;                     // W2T[e][d][h]
    W2T[i] = f2bf(W2[(e * Hn + h2) * Dn + d2]);
  }
}

// ---------------- kernel 6: fused expert FFN (WMMA bf16, all-LDS operands) ----
#define ROWS 64
#define HCK  64
#define LDA  520      // D + 8 bf16 pad
#define LDW1 520      // W1 chunk row stride (row = h, K=D contiguous)
#define LDW2 72       // W2 chunk row stride (row = d, K=HCK contiguous)
#define LDH  72       // HCK + 8
#define SWSH 36864    // max(HCK*LDW1=33280, Dn*LDW2=36864) shorts: shared W buffer

__global__ __launch_bounds__(512) void k_ffn(const float* __restrict__ x,
                                             const unsigned short* __restrict__ W1T,
                                             const unsigned short* __restrict__ W2T,
                                             const int* __restrict__ topk_idx,
                                             const float* __restrict__ topk_val,
                                             float* __restrict__ out) {
  extern __shared__ unsigned short smem[];
  unsigned short* sA = smem;                  // ROWS x LDA (gathered tokens, bf16)   66,560 B
  unsigned short* sW = sA + ROWS * LDA;       // shared W1/W2 chunk buffer            73,728 B
  unsigned short* sH = sW + SWSH;             // ROWS x LDH (GELU(h), bf16)            9,216 B
  __shared__ int   sIdx[ROWS];
  __shared__ float sVal[ROWS];

  const int b = blockIdx.z, e = blockIdx.y;
  const int rowBase = blockIdx.x * ROWS;
  const int tid = threadIdx.x;
  const int lane = tid & 31, wave = tid >> 5;
  const int half = lane >> 4, ln = lane & 15;

  if (tid < ROWS) {
    size_t off = (size_t)(b * En + e) * Cn + rowBase + tid;
    sIdx[tid] = topk_idx[off];
    sVal[tid] = topk_val[off];
  }
  __syncthreads();

  // gather selected tokens -> bf16 LDS (coalesced float4 reads + convert)
  for (int i = tid; i < ROWS * (Dn / 4); i += 512) {
    int m = i >> 7, d4 = (i & 127) * 4;
    const float4 v = *(const float4*)(x + ((size_t)(b * Tn + sIdx[m])) * Dn + d4);
    unsigned short* dst = sA + m * LDA + d4;
    dst[0] = f2bf(v.x); dst[1] = f2bf(v.y); dst[2] = f2bf(v.z); dst[3] = f2bf(v.w);
  }

  const int MT  = wave >> 2;          // out M-tile (0..3)
  const int NTb = (wave & 3) * 8;     // out N-tile base (8 tiles of 16 -> 128 cols)
  v8f acc[8];
#pragma unroll
  for (int j = 0; j < 8; ++j) acc[j] = (v8f){0.f,0.f,0.f,0.f,0.f,0.f,0.f,0.f};

  const unsigned short* w1e = W1T + ((size_t)e * Hn) * Dn;   // [h][d]
  const unsigned short* w2e = W2T + (size_t)e * Dn * Hn;     // [d][h]
  const int hm = wave >> 2, hn = wave & 3;                   // GEMM1 tile (4x4 of 16x16)

  const unsigned short* aRow  = sA + (hm * 16 + ln) * LDA  + half * 16;  // GEMM1 A row base
  const unsigned short* b1Row = sW + (hn * 16 + ln) * LDW1 + half * 16;  // GEMM1 B row base
  const unsigned short* hRow  = sH + (MT * 16 + ln) * LDH  + half * 16;  // GEMM2 A row base

  for (int h0 = 0; h0 < Hn; h0 += HCK) {
    __syncthreads();   // prev GEMM2 done reading sW/sH; sA gather done on first iter

    // ---- stage W1 chunk (async DMA): sW[n*LDW1 + k] = W1T[e][h0+n][k] ----
    for (int i = tid; i < HCK * (Dn / 8); i += 512) {
      int n = i >> 6, c = (i & 63) * 8;
      async_copy16(w1e + (size_t)(h0 + n) * Dn + c, sW + n * LDW1 + c);
    }
    async_join();
    __syncthreads();

    // ---- GEMM1 (double-buffered frags): h_tile = A(16x512) x W1chunk(512x16) ----
    v8f hacc = (v8f){0.f,0.f,0.f,0.f,0.f,0.f,0.f,0.f};
    {
      v16bf a0 = load_a_frag(aRow);
      v16bf b0 = load_b_frag(b1Row);
#pragma unroll
      for (int k0 = 0; k0 < Dn; k0 += 32) {
        if (k0 + 32 < Dn) {
          v16bf a1 = load_a_frag(aRow + k0 + 32);
          v16bf b1 = load_b_frag(b1Row + k0 + 32);
          hacc = __builtin_amdgcn_wmma_f32_16x16x32_bf16(false, a0, false, b0, (short)0, hacc, false, false);
          a0 = a1; b0 = b1;
        } else {
          hacc = __builtin_amdgcn_wmma_f32_16x16x32_bf16(false, a0, false, b0, (short)0, hacc, false, false);
        }
      }
    }
    // tanh-form GELU -> bf16 -> sH
#pragma unroll
    for (int r = 0; r < 8; ++r) {
      sH[(hm * 16 + r + half * 8) * LDH + hn * 16 + ln] = f2bf(gelu_f(hacc[r]));
    }
    __syncthreads();   // all waves done with sW (W1) and sH written

    // ---- stage W2 chunk (async DMA) into same buffer: sW[n*LDW2 + k] = W2T[e][n][h0+k] ----
    for (int i = tid; i < Dn * (HCK / 8); i += 512) {
      int n = i >> 3, c = (i & 7) * 8;
      async_copy16(w2e + (size_t)n * Hn + h0 + c, sW + n * LDW2 + c);
    }
    async_join();
    __syncthreads();

    // ---- GEMM2 (pipelined B frags): acc(64x512) += GELU(h)(64x64) x W2chunk(64x512) ----
#pragma unroll
    for (int s = 0; s < 2; ++s) {
      v16bf af = load_a_frag(hRow + s * 32);
      v16bf bcur = load_b_frag(sW + (NTb * 16 + ln) * LDW2 + s * 32 + half * 16);
#pragma unroll
      for (int j = 0; j < 8; ++j) {
        if (j < 7) {
          v16bf bnext = load_b_frag(sW + ((NTb + j + 1) * 16 + ln) * LDW2 + s * 32 + half * 16);
          acc[j] = __builtin_amdgcn_wmma_f32_16x16x32_bf16(false, af, false, bcur, (short)0, acc[j], false, false);
          bcur = bnext;
        } else {
          acc[j] = __builtin_amdgcn_wmma_f32_16x16x32_bf16(false, af, false, bcur, (short)0, acc[j], false, false);
        }
      }
    }
  }

  // weighted scatter-add epilogue
#pragma unroll
  for (int j = 0; j < 8; ++j) {
    int d = (NTb + j) * 16 + ln;
#pragma unroll
    for (int r = 0; r < 8; ++r) {
      int mloc = MT * 16 + r + half * 8;
      int t = sIdx[mloc];
      float w = sVal[mloc];
      unsafeAtomicAdd(&out[((size_t)(b * Tn + t)) * Dn + d], acc[j][r] * w);
    }
  }
}

// ---------------- kernel 7: normalize ----------------
__global__ void k_norm(float* __restrict__ out, const float* __restrict__ norm) {
  size_t stride = (size_t)gridDim.x * blockDim.x;
  for (size_t i = (size_t)blockIdx.x * blockDim.x + threadIdx.x; i < (size_t)Bn * Tn * Dn; i += stride) {
    out[i] /= fmaxf(norm[i >> 9], 1e-8f);     // i>>9 == flat token (b*T+t), D=512
  }
}

// ---------------- launch ----------------
extern "C" void kernel_launch(void* const* d_in, const int* in_sizes, int n_in,
                              void* d_out, int out_size, void* d_ws, size_t ws_size,
                              hipStream_t stream) {
  const float* x  = (const float*)d_in[0];
  const float* Wg = (const float*)d_in[1];
  const float* W1 = (const float*)d_in[2];
  const float* W2 = (const float*)d_in[3];
  float* out = (float*)d_out;

  char* ws = (char*)d_ws;
  float*          probs = (float*)(ws);                               // B*E*T f32      (1 MB)
  int*            tidx  = (int*)  (ws + 1048576);                     // B*E*C i32      (128 KB)
  float*          tval  = (float*)(ws + 1048576 + 131072);            // B*E*C f32      (128 KB)
  float*          nrm   = (float*)(ws + 1048576 + 262144);            // B*T   f32      (128 KB)
  unsigned short* W1T   = (unsigned short*)(ws + 1441792);            // E*H*D bf16     (16 MB)
  unsigned short* W2T   = (unsigned short*)(ws + 1441792 + 16777216); // E*D*H bf16     (16 MB)

  k_init<<<8192, 256, 0, stream>>>(out, nrm);
  k_gate<<<(Bn * Tn) / 8, 256, 0, stream>>>(x, Wg, probs);
  k_softmax<<<Bn * En, 1024, 0, stream>>>(probs);
  k_topc<<<Bn * En, 256, 0, stream>>>(probs, tidx, tval, nrm);
  k_convW<<<8192, 256, 0, stream>>>(W1, W2, W1T, W2T);

  const int ldsBytes = (ROWS * LDA + SWSH + ROWS * LDH) * (int)sizeof(unsigned short); // 149,504 B
  (void)hipFuncSetAttribute((const void*)k_ffn, hipFuncAttributeMaxDynamicSharedMemorySize, ldsBytes);
  k_ffn<<<dim3(Cn / ROWS, En, Bn), 512, ldsBytes, stream>>>(x, W1T, W2T, tidx, tval, out);

  k_norm<<<8192, 256, 0, stream>>>(out, nrm);
}
